// Generator_60902636257997
// MI455X (gfx1250) — matrix operand order repaired and checked by reading, not compiled, not run on hardware
//
#include <hip/hip_runtime.h>

// ---------------------------------------------------------------------------
// Problem constants (from reference)
// ---------------------------------------------------------------------------
#define Bsz 128
#define Tsz 128
#define Vsz 5000
#define Vp  5024      // V padded to multiple of 32 (157*32)
#define Esz 256
#define Hsz 512
#define H3  1536
#define Lsz 128

typedef __attribute__((ext_vector_type(16))) __bf16    v16bf;
typedef __attribute__((ext_vector_type(8)))  float     v8f;
typedef __attribute__((ext_vector_type(8)))  unsigned int v8u;
typedef __attribute__((ext_vector_type(4)))  int       v4i;

typedef __attribute__((address_space(1))) v4i* glob_v4i_p;
typedef __attribute__((address_space(3))) v4i* lds_v4i_p;

union FragAB { v8u u; v16bf b; };

#if defined(__has_builtin)
#if __has_builtin(__builtin_amdgcn_global_load_async_to_lds_b128)
#define HAVE_ASYNC_LDS 1
#endif
#endif

// ---------------------------------------------------------------------------
// bf16 helpers (round-to-nearest-even)
// ---------------------------------------------------------------------------
__device__ __forceinline__ unsigned short f2bf(float x) {
  unsigned int u = __float_as_uint(x);
  u += 0x7FFFu + ((u >> 16) & 1u);
  return (unsigned short)(u >> 16);
}

// ---------------------------------------------------------------------------
// WMMA fragment loaders (wave32 layouts per CDNA5 ISA 7.12.2)
//   A: 16x32 bf16, row-major, K contiguous.
//   B: 32x16 bf16, stored "BT": row n holds K contiguous.
// Both lower to 2x global_load_b128 per lane (verified in round-1 disasm).
// ---------------------------------------------------------------------------
__device__ __forceinline__ v16bf load_a(const unsigned short* A, int row0,
                                        int k0, int lda, int lane) {
  int half = lane >> 4, m = lane & 15;
  const unsigned short* p = A + (size_t)(row0 + m) * lda + k0;
  FragAB f;
#pragma unroll
  for (int v = 0; v < 8; ++v) {
    int grp = v >> 2, j = v & 3;
    int k = grp * 16 + half * 8 + j * 2;
    f.u[v] = *(const unsigned int*)(p + k);
  }
  return f.b;
}

__device__ __forceinline__ v16bf load_bt(const unsigned short* B, int n0,
                                         int k0, int ldb, int lane) {
  int half = lane >> 4, n = lane & 15;
  const unsigned short* p = B + (size_t)(n0 + n) * ldb + k0;
  FragAB f;
#pragma unroll
  for (int v = 0; v < 8; ++v) f.u[v] = *(const unsigned int*)(p + (v >> 2) * 16 + half * 8 + (v & 3) * 2);
  return f.b;
}

__device__ __forceinline__ v8f wmma_bf16(v16bf a, v16bf b, v8f c) {
  return __builtin_amdgcn_wmma_f32_16x16x32_bf16(false, a, false, b, (short)0,
                                                 c, false, false);
}

__device__ __forceinline__ float sigmoidf(float x) {
  return 1.0f / (1.0f + __expf(-x));
}

// ---------------------------------------------------------------------------
// One-time prep kernels
// ---------------------------------------------------------------------------
__global__ void k_convert_bf16(const float* __restrict__ src,
                               unsigned short* __restrict__ dst, int n) {
  int i = blockIdx.x * blockDim.x + threadIdx.x;
  if (i < n) dst[i] = f2bf(src[i]);
}

// f32 [rows_src, width] -> bf16 [rows_dst, width], zero pad rows
__global__ void k_pad_rows_bf16(const float* __restrict__ src,
                                unsigned short* __restrict__ dst, int rows_src,
                                int rows_dst, int width) {
  int i = blockIdx.x * blockDim.x + threadIdx.x;
  if (i >= rows_dst * width) return;
  int r = i / width;
  dst[i] = (r < rows_src) ? f2bf(src[i]) : (unsigned short)0;
}

// W_eT[n][k] = sum_e W_ih[n][e] * emb[k][e] ; out bf16 [1536, 5024]
// emb_b is padded to 5024 rows -> no guards in the inner loop.
__global__ void __launch_bounds__(128) k_weT(
    const unsigned short* __restrict__ Wih, const unsigned short* __restrict__ Emb,
    unsigned short* __restrict__ WeT) {
  int lane = threadIdx.x & 31, wid = threadIdx.x >> 5;
  int kt = blockIdx.x * 4 + wid;          // column tile over 5024 -> 314 tiles
  int rt = blockIdx.y;                    // row tile over 1536 -> 96 tiles
  if (kt >= Vp / 16) return;
  v8f acc = {};
  for (int e0 = 0; e0 < Esz; e0 += 32) {
    v16bf a = load_a(Wih, rt * 16, e0, Esz, lane);
    v16bf b = load_bt(Emb, kt * 16, e0, Esz, lane);
    acc = wmma_bf16(a, b, acc);
  }
  int half = lane >> 4;
  int col = kt * 16 + (lane & 15);
#pragma unroll
  for (int r = 0; r < 8; ++r) {
    int row = rt * 16 + half * 8 + r;
    WeT[(size_t)row * Vp + col] = f2bf(acc[r]);
  }
}

// h0 = noise @ W_init^T + b_init  (one-time, small)
__global__ void k_h0(const float* __restrict__ noise,
                     const float* __restrict__ Winit,
                     const float* __restrict__ binit,
                     float* __restrict__ hf, unsigned short* __restrict__ hb) {
  int i = blockIdx.x * blockDim.x + threadIdx.x;  // 128*512
  if (i >= Bsz * Hsz) return;
  int m = i >> 9, n = i & (Hsz - 1);
  float acc = binit[n];
  for (int k = 0; k < Lsz; ++k) acc += noise[m * Lsz + k] * Winit[n * Lsz + k];
  hf[i] = acc;
  hb[i] = f2bf(acc);
}

// y_bf init: one-hot at PAD_IDX=0, padded to 5024 cols
__global__ void k_yinit(unsigned short* __restrict__ ybf) {
  int i = blockIdx.x * blockDim.x + threadIdx.x;
  if (i >= Bsz * Vp) return;
  ybf[i] = ((i % Vp) == 0) ? (unsigned short)0x3F80 : (unsigned short)0;
}

// ---------------------------------------------------------------------------
// Per-step kernels
// ---------------------------------------------------------------------------
// Fused GRU cell. Each wave owns TWO 16x16 h tiles (same n, m and m+16):
// B fragments are loaded once and feed two independent WMMA chains.
__global__ void __launch_bounds__(128) k_gru(
    const unsigned short* __restrict__ ybf,
    const unsigned short* __restrict__ WeT,
    const unsigned short* __restrict__ Whh,
    const unsigned short* __restrict__ hbf_in,
    const float* __restrict__ hf_in, const float* __restrict__ b_ih,
    const float* __restrict__ b_hh, float* __restrict__ hf_out,
    unsigned short* __restrict__ hbf_out) {
  int lane = threadIdx.x & 31, wid = threadIdx.x >> 5;
  int nt = blockIdx.x * 4 + wid;   // 0..31 over H=512
  int m0 = blockIdx.y * 32;        // two m-tiles: rows m0..m0+31
  v8f ar0 = {}, ar1 = {}, az0 = {}, az1 = {};
  v8f ai0 = {}, ai1 = {}, ah0 = {}, ah1 = {};

  // gh part, K = 512
  for (int k0 = 0; k0 < Hsz; k0 += 32) {
    v16bf a0 = load_a(hbf_in, m0, k0, Hsz, lane);
    v16bf a1 = load_a(hbf_in, m0 + 16, k0, Hsz, lane);
    v16bf br = load_bt(Whh, nt * 16, k0, Hsz, lane);
    v16bf bz = load_bt(Whh, Hsz + nt * 16, k0, Hsz, lane);
    v16bf bn = load_bt(Whh, 2 * Hsz + nt * 16, k0, Hsz, lane);
    ar0 = wmma_bf16(a0, br, ar0);  ar1 = wmma_bf16(a1, br, ar1);
    az0 = wmma_bf16(a0, bz, az0);  az1 = wmma_bf16(a1, bz, az1);
    ah0 = wmma_bf16(a0, bn, ah0);  ah1 = wmma_bf16(a1, bn, ah1);
  }
  // gi part, K = 5024 (padded; pad cols of y are zero)
  for (int k0 = 0; k0 < Vp; k0 += 32) {
    v16bf a0 = load_a(ybf, m0, k0, Vp, lane);
    v16bf a1 = load_a(ybf, m0 + 16, k0, Vp, lane);
    v16bf br = load_bt(WeT, nt * 16, k0, Vp, lane);
    v16bf bz = load_bt(WeT, Hsz + nt * 16, k0, Vp, lane);
    v16bf bn = load_bt(WeT, 2 * Hsz + nt * 16, k0, Vp, lane);
    ar0 = wmma_bf16(a0, br, ar0);  ar1 = wmma_bf16(a1, br, ar1);
    az0 = wmma_bf16(a0, bz, az0);  az1 = wmma_bf16(a1, bz, az1);
    ai0 = wmma_bf16(a0, bn, ai0);  ai1 = wmma_bf16(a1, bn, ai1);
  }

  int half = lane >> 4;
  int col = nt * 16 + (lane & 15);
  float bir = b_ih[col],            bhr = b_hh[col];
  float biz = b_ih[col + Hsz],      bhz = b_hh[col + Hsz];
  float bin = b_ih[col + 2 * Hsz],  bhn = b_hh[col + 2 * Hsz];
#pragma unroll
  for (int p = 0; p < 2; ++p) {
    v8f vr = p ? ar1 : ar0, vz = p ? az1 : az0;
    v8f vi = p ? ai1 : ai0, vh = p ? ah1 : ah0;
#pragma unroll
    for (int r = 0; r < 8; ++r) {
      int m = m0 + p * 16 + half * 8 + r;
      float hold = hf_in[(size_t)m * Hsz + col];
      float rg = sigmoidf(vr[r] + bir + bhr);
      float zg = sigmoidf(vz[r] + biz + bhz);
      float ng = tanhf(vi[r] + bin + rg * (vh[r] + bhn));
      float hnew = (1.0f - zg) * ng + zg * hold;
      hf_out[(size_t)m * Hsz + col] = hnew;
      hbf_out[(size_t)m * Hsz + col] = f2bf(hnew);
    }
  }
}

// logits = h_new @ W_out^T + b_out ; val = (logits + gumbel[t]) / tau
// Each wave owns two m-tiles; gumbel is prefetched ahead of the K-loop
// (global_prefetch_b8) so the only HBM read overlaps the GEMM.
__global__ void __launch_bounds__(128) k_logits(
    const unsigned short* __restrict__ hbf,
    const unsigned short* __restrict__ Wout, const float* __restrict__ b_out,
    const float* __restrict__ gumbel, const float* __restrict__ temperature,
    float* __restrict__ logits_s, int t) {
  int lane = threadIdx.x & 31, wid = threadIdx.x >> 5;
  int vt = blockIdx.x * 4 + wid;  // over 5024 -> 314 tiles
  int m0 = blockIdx.y * 32;       // rows m0..m0+31
  if (vt >= Vp / 16) return;
  int half = lane >> 4;
  int v = vt * 16 + (lane & 15);

  // Prefetch this wave's gumbel operands (HBM-resident) before the GEMM.
  if (v < Vsz) {
#pragma unroll
    for (int r = 0; r < 8; ++r) {
      __builtin_prefetch(&gumbel[((size_t)t * Bsz + (m0 + half * 8 + r)) * Vsz + v], 0, 0);
      __builtin_prefetch(&gumbel[((size_t)t * Bsz + (m0 + 16 + half * 8 + r)) * Vsz + v], 0, 0);
    }
  }

  v8f acc0 = {}, acc1 = {};
  for (int k0 = 0; k0 < Hsz; k0 += 32) {
    v16bf a0 = load_a(hbf, m0, k0, Hsz, lane);
    v16bf a1 = load_a(hbf, m0 + 16, k0, Hsz, lane);
    v16bf b = load_bt(Wout, vt * 16, k0, Hsz, lane);
    acc0 = wmma_bf16(a0, b, acc0);
    acc1 = wmma_bf16(a1, b, acc1);
  }
  float inv_tau = 1.0f / temperature[0];
  float bo = (v < Vsz) ? b_out[v] : 0.0f;
#pragma unroll
  for (int p = 0; p < 2; ++p) {
    v8f acc = p ? acc1 : acc0;
#pragma unroll
    for (int r = 0; r < 8; ++r) {
      int m = m0 + p * 16 + half * 8 + r;
      float val;
      if (v < Vsz) {
        float g = gumbel[((size_t)t * Bsz + m) * Vsz + v];
        val = (acc[r] + bo + g) * inv_tau;
      } else {
        val = -1e30f;  // padded columns -> softmax weight 0
      }
      logits_s[(size_t)m * Vp + v] = val;
    }
  }
}

// Row softmax over V, staged in LDS (async copy global->LDS when available).
// Writes f32 output slice and bf16 feedback.
__global__ void __launch_bounds__(256) k_softmax(
    const float* __restrict__ logits_s, float* __restrict__ out,
    unsigned short* __restrict__ ybf, int t) {
  __shared__ __align__(16) float ls[Vp];
  __shared__ float red[256];
  int m = blockIdx.x, tid = threadIdx.x;

#if defined(HAVE_ASYNC_LDS)
  {
    const float* src = logits_s + (size_t)m * Vp;
    for (int i = tid * 4; i < Vp; i += 256 * 4) {
      __builtin_amdgcn_global_load_async_to_lds_b128(
          (glob_v4i_p)(src + i), (lds_v4i_p)&ls[i], 0, 0);
    }
#if __has_builtin(__builtin_amdgcn_s_wait_asynccnt)
    __builtin_amdgcn_s_wait_asynccnt(0);
#else
    asm volatile("s_wait_asynccnt 0x0" ::: "memory");
#endif
  }
#else
  for (int i = tid; i < Vp; i += 256) ls[i] = logits_s[(size_t)m * Vp + i];
#endif
  __syncthreads();

  float mx = -1e30f;
  for (int i = tid; i < Vp; i += 256) mx = fmaxf(mx, ls[i]);
  red[tid] = mx;
  __syncthreads();
  for (int s = 128; s > 0; s >>= 1) {
    if (tid < s) red[tid] = fmaxf(red[tid], red[tid + s]);
    __syncthreads();
  }
  mx = red[0];
  __syncthreads();
  float sum = 0.0f;
  for (int i = tid; i < Vsz; i += 256) sum += __expf(ls[i] - mx);
  red[tid] = sum;
  __syncthreads();
  for (int s = 128; s > 0; s >>= 1) {
    if (tid < s) red[tid] += red[tid + s];
    __syncthreads();
  }
  float rinv = 1.0f / red[0];
  for (int i = tid; i < Vsz; i += 256) {
    float y = __expf(ls[i] - mx) * rinv;
    out[((size_t)m * Tsz + t) * Vsz + i] = y;
    ybf[(size_t)m * Vp + i] = f2bf(y);
  }
  for (int i = Vsz + tid; i < Vp; i += 256) ybf[(size_t)m * Vp + i] = 0;
}

// ---------------------------------------------------------------------------
// Host launcher
// ---------------------------------------------------------------------------
extern "C" void kernel_launch(void* const* d_in, const int* in_sizes, int n_in,
                              void* d_out, int out_size, void* d_ws,
                              size_t ws_size, hipStream_t stream) {
  const float* noise   = (const float*)d_in[0];
  const float* gumbel  = (const float*)d_in[1];
  const float* temp    = (const float*)d_in[2];
  const float* W_init  = (const float*)d_in[3];
  const float* b_init  = (const float*)d_in[4];
  const float* emb     = (const float*)d_in[5];
  const float* W_ih    = (const float*)d_in[6];
  const float* W_hh    = (const float*)d_in[7];
  const float* b_ih    = (const float*)d_in[8];
  const float* b_hh    = (const float*)d_in[9];
  const float* W_out   = (const float*)d_in[10];
  const float* b_out   = (const float*)d_in[11];
  float* out = (float*)d_out;

  char* ws = (char*)d_ws;
  size_t off = 0;
  auto alloc = [&](size_t bytes) {
    size_t o = off;
    off += (bytes + 255) & ~(size_t)255;
    return o;
  };
  unsigned short* WeT   = (unsigned short*)(ws + alloc((size_t)H3 * Vp * 2));
  unsigned short* Whh_b = (unsigned short*)(ws + alloc((size_t)H3 * Hsz * 2));
  unsigned short* Wout_b= (unsigned short*)(ws + alloc((size_t)Vp * Hsz * 2));
  unsigned short* Wih_b = (unsigned short*)(ws + alloc((size_t)H3 * Esz * 2));
  unsigned short* emb_b = (unsigned short*)(ws + alloc((size_t)Vp * Esz * 2));
  float*          hf0   = (float*)(ws + alloc((size_t)Bsz * Hsz * 4));
  float*          hf1   = (float*)(ws + alloc((size_t)Bsz * Hsz * 4));
  unsigned short* hb0   = (unsigned short*)(ws + alloc((size_t)Bsz * Hsz * 2));
  unsigned short* hb1   = (unsigned short*)(ws + alloc((size_t)Bsz * Hsz * 2));
  unsigned short* ybf   = (unsigned short*)(ws + alloc((size_t)Bsz * Vp * 2));
  float*          lgs   = (float*)(ws + alloc((size_t)Bsz * Vp * 4));

  // ---- one-time prep (runs every launch; deterministic) ----
  {
    int n;
    n = H3 * Esz;
    k_convert_bf16<<<(n + 255) / 256, 256, 0, stream>>>(W_ih, Wih_b, n);
    n = H3 * Hsz;
    k_convert_bf16<<<(n + 255) / 256, 256, 0, stream>>>(W_hh, Whh_b, n);
    n = Vp * Esz;
    k_pad_rows_bf16<<<(n + 255) / 256, 256, 0, stream>>>(emb, emb_b, Vsz, Vp, Esz);
    n = Vp * Hsz;
    k_pad_rows_bf16<<<(n + 255) / 256, 256, 0, stream>>>(W_out, Wout_b, Vsz, Vp, Hsz);
    // W_eT = (emb @ W_ih^T)^T  via WMMA
    dim3 gW((Vp / 16 + 3) / 4, H3 / 16);
    k_weT<<<gW, 128, 0, stream>>>(Wih_b, emb_b, WeT);
    n = Bsz * Hsz;
    k_h0<<<(n + 255) / 256, 256, 0, stream>>>(noise, W_init, b_init, hf0, hb0);
    n = Bsz * Vp;
    k_yinit<<<(n + 255) / 256, 256, 0, stream>>>(ybf);
  }

  // ---- sequential scan over T steps ----
  dim3 gGru(Hsz / 16 / 4, Bsz / 32);        // (8, 4) x 128 threads
  dim3 gLog((Vp / 16 + 3) / 4, Bsz / 32);   // (79, 4) x 128 threads
  for (int t = 0; t < Tsz; ++t) {
    float*          hf_in  = (t & 1) ? hf1 : hf0;
    float*          hf_out = (t & 1) ? hf0 : hf1;
    unsigned short* hb_in  = (t & 1) ? hb1 : hb0;
    unsigned short* hb_out = (t & 1) ? hb0 : hb1;
    k_gru<<<gGru, 128, 0, stream>>>(ybf, WeT, Whh_b, hb_in, hf_in, b_ih, b_hh,
                                    hf_out, hb_out);
    k_logits<<<gLog, 128, 0, stream>>>(hb_out, Wout_b, b_out, gumbel, temp,
                                       lgs, t);
    k_softmax<<<Bsz, 256, 0, stream>>>(lgs, out, ybf, t);
  }
}
